// BiLSTM_CRF_77532749627592
// MI455X (gfx1250) — matrix operand order repaired and checked by reading, not compiled, not run on hardware
//
#include <hip/hip_runtime.h>
#include <hip/hip_bf16.h>

// ---------------- problem constants ----------------
#define T_   4096
#define E_   512
#define H_   256
#define G4H  1024   // 4*H gates per direction
#define K_   12
#define NEG_ (-10000.0f)

typedef __attribute__((ext_vector_type(16))) _Float16 v16h;
typedef __attribute__((ext_vector_type(8)))  float    v8f;

// =====================================================================
// K0a: embedding gather + f32->f16 convert.  Xh[t][e] = emb[sent[t]][e]
// =====================================================================
__global__ void k_embed(const int* __restrict__ sent,
                        const float* __restrict__ emb,
                        _Float16* __restrict__ Xh) {
    int idx = blockIdx.x * blockDim.x + threadIdx.x;       // < T_*E_
    int t = idx >> 9;          // /512
    int e = idx & (E_ - 1);
    size_t src = (size_t)sent[t] * E_ + e;
    Xh[idx] = (_Float16)emb[src];
}

// =====================================================================
// K0b: pre-swizzle W_ih into WMMA B-fragment order (f16), both dirs.
//  Fragment for (nt, kt, lane): 16 halves  B[k0+i][n] = w_ih[n][k0+i]
//  with n = nt*16 + (lane&15), k0 = kt*32 + (lane<16 ? 0 : 16).
//  Source reads are 16 contiguous f32 (w_ih row-major in K) -> coalesced.
//  GEMM then loads each fragment as 2 contiguous 16B loads per lane.
// =====================================================================
__global__ void k_bswz(const float* __restrict__ wf, const float* __restrict__ wb,
                       _Float16* __restrict__ of, _Float16* __restrict__ ob) {
    int idx = blockIdx.x * blockDim.x + threadIdx.x;   // < 2 * 64*16*32
    const int per = 64 * 16 * 32;                      // 32768 fragments/dir
    int d = idx >= per;
    int r = idx - d * per;
    int lane = r & 31;
    int rt   = r >> 5;        // nt*16 + kt
    int kt   = rt & 15;
    int nt   = rt >> 4;
    const float* w = d ? wb : wf;
    _Float16*    o = (d ? ob : of) + (size_t)r * 16;
    int n  = nt * 16 + (lane & 15);
    int k0 = kt * 32 + ((lane < 16) ? 0 : 16);
    const float* src = w + (size_t)n * E_ + k0;
    #pragma unroll
    for (int i = 0; i < 16; ++i) o[i] = (_Float16)src[i];
}

// =====================================================================
// K0c: W_hh f32 -> f16 (row-major kept: [4H][H]) both directions
// =====================================================================
__global__ void k_whh16(const float* __restrict__ wf, const float* __restrict__ wb,
                        _Float16* __restrict__ of, _Float16* __restrict__ ob) {
    int idx = blockIdx.x * blockDim.x + threadIdx.x;       // < 2*G4H*H_
    const int per = G4H * H_;
    int d = idx >= per;
    int r = idx - d * per;
    const float* w = d ? wb : wf;
    _Float16*    o = d ? ob : of;
    o[r] = (_Float16)w[r];
}

// =====================================================================
// K1: WMMA GEMM  G_d = X_d @ W_ih_dT + b_d     (f16 in, f32 out)
//  one 16x16 C tile per wave; K loop of 16 x v_wmma_f32_16x16x32_f16.
//  A loaded straight from Xh (2x16B/lane), B from pre-swizzled fragments
//  (2x16B/lane, coalesced). Backward direction reads time-reversed A rows.
// =====================================================================
__global__ void __launch_bounds__(256)
k_gemm(const _Float16* __restrict__ Xh,
       const _Float16* __restrict__ Bswf, const _Float16* __restrict__ Bswb,
       const float* __restrict__ bf,      const float* __restrict__ bb,
       float* __restrict__ Gf,            float* __restrict__ Gb) {
    int wave = threadIdx.x >> 5;
    int lane = threadIdx.x & 31;
    int ln   = lane & 15;
    bool hi  = lane >= 16;

    int g  = blockIdx.x * 8 + wave;      // global tile id, [0, 32768)
    int d  = g >> 14;                    // 16384 tiles per direction
    int r  = g & 16383;
    int mt = r >> 6;                     // 256 M tiles
    int nt = r & 63;                     // 64  N tiles

    const _Float16* Bsw  = d ? Bswb : Bswf;
    const float*    bias = d ? bb   : bf;
    float*          G    = d ? Gb   : Gf;

    int rowA = mt * 16 + ln;
    int t    = d ? (T_ - 1 - rowA) : rowA;      // backward dir: reversed time
    const _Float16* arow = Xh + (size_t)t * E_;
    int ncol = nt * 16 + ln;

    v8f acc = {0.f, 0.f, 0.f, 0.f, 0.f, 0.f, 0.f, 0.f};

    #pragma unroll 4
    for (int kt = 0; kt < 16; ++kt) {
        // A fragment: lane<16 holds K {0..15}, lane>=16 holds {16..31}
        union { uint4 u[2]; v16h v; } a;
        const _Float16* ap = arow + kt * 32 + (hi ? 16 : 0);
        a.u[0] = *(const uint4*)(ap);
        a.u[1] = *(const uint4*)(ap + 8);

        // B fragment: pre-swizzled, contiguous 32B per lane
        union { uint4 u[2]; v16h v; } b;
        const _Float16* bp = Bsw + ((size_t)((nt * 16 + kt) * 32 + lane) << 4);
        b.u[0] = *(const uint4*)(bp);
        b.u[1] = *(const uint4*)(bp + 8);

        acc = __builtin_amdgcn_wmma_f32_16x16x32_f16(
                  false, a.v, false, b.v, (short)0, acc, false, false);
    }

    float bv    = bias[ncol];
    int   mbase = mt * 16 + (hi ? 8 : 0);
    #pragma unroll
    for (int rr = 0; rr < 8; ++rr)
        G[(size_t)(mbase + rr) * G4H + ncol] = acc[rr] + bv;
}

// =====================================================================
// K2: sequential LSTM recurrence, one workgroup (32 waves) per direction.
//  W_hh (f16, 512KB) lives PERSISTENTLY in WMMA A-fragments: 16 x v16h
//  per wave (= 128 VGPRs/lane across the WGP). Per step:
//   phase 0: h broadcast into all 16 B columns (32B LDS loads), 16 WMMAs
//   phase 1: publish 1024 gate sums (4 x ds_store_b128 per wave)
//   phase 2: ALL 1024 threads apply their gate nonlinearity (TRANS work
//            spread over all 4 SIMDs; region branch is wave-uniform)
//   phase 3: threads 0..255 combine into c/h, write h16 + global h
// =====================================================================
__global__ void __launch_bounds__(1024)
k_recur(const _Float16* __restrict__ Wh16f, const _Float16* __restrict__ Wh16b,
        const float* __restrict__ Gf,       const float* __restrict__ Gb,
        float* __restrict__ hf,             float* __restrict__ hb) {
    int dir = blockIdx.x;
    const _Float16* W = dir ? Wh16b : Wh16f;
    const float*    G = dir ? Gb    : Gf;
    float*       hout = dir ? hb    : hf;

    int tid  = threadIdx.x;
    int wave = tid >> 5;
    int lane = tid & 31;
    int ln   = lane & 15;
    bool hi  = lane >= 16;

    __shared__ __attribute__((aligned(32))) _Float16 h16[H_];
    __shared__ __attribute__((aligned(16))) float gates[G4H];
    __shared__ float actv[G4H];

    // ---- load persistent A fragments of W_hh: 2 M-tiles x 8 K-tiles ----
    v16h afrag[2][8];
    #pragma unroll
    for (int j = 0; j < 2; ++j) {
        int row = (2 * wave + j) * 16 + ln;
        #pragma unroll
        for (int kt = 0; kt < 8; ++kt) {
            union { uint4 u[2]; v16h v; } a;
            const _Float16* ap = W + (size_t)row * H_ + kt * 32 + (hi ? 16 : 0);
            a.u[0] = *(const uint4*)(ap);
            a.u[1] = *(const uint4*)(ap + 8);
            afrag[j][kt] = a.v;
        }
    }

    float c = 0.f;
    if (tid < H_) h16[tid] = (_Float16)0.f;   // h_0 = 0

    for (int t = 0; t < T_; ++t) {
        __syncthreads();   // h16 ready; previous gates/actv readers done

        // ---- phase 0: matvec W_hh @ h via WMMA (h replicated over N) ----
        v8f acc0 = {0.f, 0.f, 0.f, 0.f, 0.f, 0.f, 0.f, 0.f};
        v8f acc1 = {0.f, 0.f, 0.f, 0.f, 0.f, 0.f, 0.f, 0.f};
        #pragma unroll
        for (int kt = 0; kt < 8; ++kt) {
            v16h b = *(const v16h*)(h16 + kt * 32 + (hi ? 16 : 0));
            acc0 = __builtin_amdgcn_wmma_f32_16x16x32_f16(
                       false, afrag[0][kt], false, b, (short)0, acc0, false, false);
            acc1 = __builtin_amdgcn_wmma_f32_16x16x32_f16(
                       false, afrag[1][kt], false, b, (short)0, acc1, false, false);
        }

        // ---- phase 1: publish gate sums (columns identical; lanes 0/16) ----
        if (ln == 0) {
            int mb = hi ? 8 : 0;
            float4* g0 = (float4*)&gates[(2 * wave + 0) * 16 + mb];
            float4* g1 = (float4*)&gates[(2 * wave + 1) * 16 + mb];
            g0[0] = make_float4(acc0[0], acc0[1], acc0[2], acc0[3]);
            g0[1] = make_float4(acc0[4], acc0[5], acc0[6], acc0[7]);
            g1[0] = make_float4(acc1[0], acc1[1], acc1[2], acc1[3]);
            g1[1] = make_float4(acc1[4], acc1[5], acc1[6], acc1[7]);
        }
        __syncthreads();

        // ---- phase 2: all 1024 threads activate their gate ----
        {
            const float* g = G + (size_t)t * G4H;
            float v = g[tid] + gates[tid];
            float a;
            if (tid < 2 * H_ || tid >= 3 * H_)        // i, f, o gates
                a = 1.f / (1.f + expf(-v));
            else                                       // g (cell) gate
                a = tanhf(v);
            actv[tid] = a;
            if (t + 1 < T_)                            // warm next preact row
                __builtin_prefetch(&G[(size_t)(t + 1) * G4H + tid], 0, 1);
        }
        __syncthreads();

        // ---- phase 3: combine into cell/hidden state ----
        if (tid < H_) {
            c = actv[H_ + tid] * c + actv[tid] * actv[2 * H_ + tid];
            float h = actv[3 * H_ + tid] * tanhf(c);
            h16[tid] = (_Float16)h;
            int ts = dir ? (T_ - 1 - t) : t;   // un-reverse backward direction
            hout[(size_t)ts * H_ + tid] = h;
        }
    }
}

// =====================================================================
// K3: emission scores  feats[t][k] = w_tag[k]·[h_f[t];h_b[t]] + b_tag[k]
// =====================================================================
__global__ void k_feats(const float* __restrict__ hf, const float* __restrict__ hb,
                        const float* __restrict__ wtag, const float* __restrict__ btag,
                        float* __restrict__ feats) {
    int idx = blockIdx.x * blockDim.x + threadIdx.x;   // < T_*K_
    if (idx >= T_ * K_) return;
    int t = idx / K_;
    int k = idx - t * K_;
    const float* w = wtag + k * (2 * H_);
    const float* hfr = hf + (size_t)t * H_;
    const float* hbr = hb + (size_t)t * H_;
    float s = btag[k];
    for (int j = 0; j < H_; ++j) s += w[j]       * hfr[j];
    for (int j = 0; j < H_; ++j) s += w[H_ + j]  * hbr[j];
    feats[idx] = s;
}

// =====================================================================
// K4: CRF forward algorithm (sequential over T), single wave, K=12 lanes.
// =====================================================================
__global__ void k_crf(const float* __restrict__ feats,
                      const float* __restrict__ trans,
                      float* __restrict__ out) {
    __shared__ float alpha[K_];
    int k = threadIdx.x;

    float tr[K_];
    if (k < K_) {
        #pragma unroll
        for (int p = 0; p < K_; ++p) tr[p] = trans[k * K_ + p];  // score p -> k
        alpha[k] = (k == K_ - 2) ? 0.f : NEG_;                    // START = K-2
    }
    __syncthreads();

    for (int t = 0; t < T_; ++t) {
        float nv = 0.f;
        if (k < K_) {
            float a[K_], m = -3.4e38f;
            #pragma unroll
            for (int p = 0; p < K_; ++p) { a[p] = alpha[p] + tr[p]; m = fmaxf(m, a[p]); }
            float s = 0.f;
            #pragma unroll
            for (int p = 0; p < K_; ++p) s += expf(a[p] - m);
            nv = m + logf(s) + feats[t * K_ + k];
        }
        __syncthreads();
        if (k < K_) alpha[k] = nv;
        __syncthreads();
    }

    if (k == 0) {
        float a[K_], m = -3.4e38f;
        #pragma unroll
        for (int p = 0; p < K_; ++p) {
            a[p] = alpha[p] + trans[(K_ - 1) * K_ + p];   // END = K-1
            m = fmaxf(m, a[p]);
        }
        float s = 0.f;
        #pragma unroll
        for (int p = 0; p < K_; ++p) s += expf(a[p] - m);
        out[0] = m + logf(s);
    }
}

// =====================================================================
// host-side launch
// =====================================================================
extern "C" void kernel_launch(void* const* d_in, const int* in_sizes, int n_in,
                              void* d_out, int out_size, void* d_ws, size_t ws_size,
                              hipStream_t stream) {
    const int*   sentence    = (const int*)  d_in[0];
    const float* emb_table   = (const float*)d_in[1];
    const float* w_ih_f      = (const float*)d_in[2];
    const float* w_hh_f      = (const float*)d_in[3];
    const float* b_f         = (const float*)d_in[4];
    const float* w_ih_b      = (const float*)d_in[5];
    const float* w_hh_b      = (const float*)d_in[6];
    const float* b_b         = (const float*)d_in[7];
    const float* w_tag       = (const float*)d_in[8];
    const float* b_tag       = (const float*)d_in[9];
    const float* transitions = (const float*)d_in[10];

    // ---- carve workspace (all regions 256B aligned; total ~47.3 MB) ----
    char* p = (char*)d_ws;
    auto alloc = [&](size_t bytes) -> char* {
        char* r = p;
        p += (bytes + 255) & ~(size_t)255;
        return r;
    };
    _Float16* Xh    = (_Float16*)alloc((size_t)T_ * E_ * 2);        // 4 MB
    _Float16* Bswf  = (_Float16*)alloc((size_t)E_ * G4H * 2);       // 1 MB (swizzled)
    _Float16* Bswb  = (_Float16*)alloc((size_t)E_ * G4H * 2);       // 1 MB
    _Float16* Wh16f = (_Float16*)alloc((size_t)G4H * H_ * 2);       // 0.5 MB
    _Float16* Wh16b = (_Float16*)alloc((size_t)G4H * H_ * 2);       // 0.5 MB
    float*    Gf    = (float*)   alloc((size_t)T_ * G4H * 4);       // 16 MB
    float*    Gb    = (float*)   alloc((size_t)T_ * G4H * 4);       // 16 MB
    float*    hf    = (float*)   alloc((size_t)T_ * H_ * 4);        // 4 MB
    float*    hb    = (float*)   alloc((size_t)T_ * H_ * 4);        // 4 MB
    float*    feats = (float*)   alloc((size_t)T_ * K_ * 4);        // 192 KB

    // ---- pipeline (stream-ordered: dependencies are implicit) ----
    k_embed <<<(T_ * E_) / 256,        256, 0, stream>>>(sentence, emb_table, Xh);
    k_bswz  <<<(2 * 64 * 16 * 32) / 256, 256, 0, stream>>>(w_ih_f, w_ih_b, Bswf, Bswb);
    k_whh16 <<<(2 * G4H * H_) / 256,   256, 0, stream>>>(w_hh_f, w_hh_b, Wh16f, Wh16b);

    // 32768 tiles (both dirs), 8 waves/block -> 4096 blocks
    k_gemm  <<<4096, 256, 0, stream>>>(Xh, Bswf, Bswb, b_f, b_b, Gf, Gb);

    // one 32-wave workgroup per direction; W_hh persistent in VGPR A-fragments
    k_recur <<<2, 1024, 0, stream>>>(Wh16f, Wh16b, Gf, Gb, hf, hb);

    k_feats <<<(T_ * K_ + 255) / 256, 256, 0, stream>>>(hf, hb, w_tag, b_tag, feats);
    k_crf   <<<1, 32, 0, stream>>>(feats, transitions, (float*)d_out);
}
